// ProjectionHead_89678917140570
// MI455X (gfx1250) — matrix operand order repaired and checked by reading, not compiled, not run on hardware
//
#include <hip/hip_runtime.h>
#include <hip/hip_bf16.h>

typedef __attribute__((ext_vector_type(16))) _Float16 v16h;
typedef __attribute__((ext_vector_type(8)))  _Float16 v8h;
typedef __attribute__((ext_vector_type(8)))  float    v8f;
typedef __attribute__((ext_vector_type(4)))  unsigned int u32x4;
typedef __attribute__((ext_vector_type(8)))  unsigned int u32x8;

#define MB   32          // positions per block tile
#define LDA  264         // halves per LDS row (528B = 33*16B -> b128-aligned)
#define HWSZ 16384       // H*W

// ---------------------------------------------------------------------------
// Weight pre-convert: W1 (256x256) and W2 (128x256) f32 -> f16 (done once,
// removes per-k-step v_cvt from the GEMM hot loops and halves B-operand bytes)
// ---------------------------------------------------------------------------
__global__ __launch_bounds__(256) void cvt_kernel(
    const float* __restrict__ W1, const float* __restrict__ W2,
    _Float16* __restrict__ W1h, _Float16* __restrict__ W2h)
{
    const int i = blockIdx.x * 256 + threadIdx.x;
    if (i < 256 * 256) W1h[i] = (_Float16)W1[i];
    if (i < 128 * 256) W2h[i] = (_Float16)W2[i];
}

// ---------------------------------------------------------------------------
// Fused projector: z = normalize( relu(x@W1^T + b1) @ W2^T + b2 ), per position
// W2 (f16, 64KB) is DMA'd into LDS once per block via the Tensor Data Mover,
// overlapped with the X-tile staging; GEMM2 B-frags then come from LDS.
// ---------------------------------------------------------------------------
__global__ __launch_bounds__(256) void proj_kernel(
    const float*    __restrict__ x,
    const _Float16* __restrict__ W1h, const float* __restrict__ b1,
    const _Float16* __restrict__ W2h, const float* __restrict__ b2,
    float* __restrict__ z)
{
    __shared__ __align__(16) _Float16 A_s[MB * LDA];     // x tile, f16
    __shared__ __align__(16) _Float16 H_s[MB * LDA];     // hidden tile, f16
    __shared__ __align__(16) float    Z_s[MB * 128];     // output tile, f32
    __shared__ __align__(16) _Float16 W2_s[128 * 256];   // full W2 f16 (64KB)
    __shared__ float rnorm_s[MB];

    const int t     = threadIdx.x;
    const int lane  = t & 31;
    const int wave  = t >> 5;          // 0..7
    const int mlo   = lane & 15;       // row/col within 16
    const int hi    = lane >> 4;       // lane group 0/1
    const int kbA   = hi * 16;         // A-frag k-base
    const int kbB   = hi * 16;         // B-frag k-base
    const int base  = blockIdx.x * MB; // first global position

    // ---- kick off TDM: W2h (global, f16, 32768 elems) -> W2_s (LDS) ----
    if (wave == 0) {
        const unsigned long long ga = (unsigned long long)W2h;
        const unsigned int lds_off  = (unsigned int)(unsigned long long)&W2_s[0];
        u32x4 g0;
        g0[0] = 1u;                                   // count=1, user mode
        g0[1] = lds_off;                              // lds_addr (bytes)
        g0[2] = (unsigned int)(ga & 0xFFFFFFFFu);     // global_addr[31:0]
        g0[3] = (unsigned int)((ga >> 32) & 0x1FFFFFFu) | (2u << 30); // [56:32] | type=2
        u32x8 g1 = {};
        g1[0] = (1u << 16);          // data_size=1 (2 bytes)
        g1[1] = (32768u << 16);      // tensor_dim0[15:0] = 32768
        g1[2] = (1u << 16);          // tensor_dim0[31:16]=0 | tensor_dim1[15:0]=1
        g1[3] = (32768u << 16);      // tensor_dim1[31:16]=0 | tile_dim0 = 32768
        g1[4] = 1u;                  // tile_dim1 = 1, tile_dim2 = 0
        g1[5] = 32768u;              // tensor_dim0_stride[31:0]
        g1[6] = 0u;                  // stride[47:32] | tensor_dim1_stride[15:0]
        g1[7] = 0u;                  // tensor_dim1_stride[47:16]
        u32x4 g2 = {};
        u32x4 g3 = {};
        asm volatile("tensor_load_to_lds %0, %1, %2, %3"
                     :: "s"(g0), "s"(g1), "s"(g2), "s"(g3) : "memory");
    }

    // ---- stage X tile (MB x 256) -> LDS f16; lanes contiguous in position ----
    for (int i = t; i < MB * 256; i += 256) {
        int m  = i & (MB - 1);
        int c  = i / MB;
        int p  = base + m;
        int b  = p >> 14;
        int hw = p & (HWSZ - 1);
        float v = x[((size_t)(b * 256 + c) << 14) + hw];
        A_s[m * LDA + c] = (_Float16)v;
    }
    if (wave == 0) __builtin_amdgcn_s_wait_tensorcnt(0);
    __syncthreads();

    // ---- GEMM1: (MB x 256) x (256 x 256) + b1, ReLU -> H_s (f16) ----
    // 2 Mtiles x 16 Ntiles = 32 WMMA tiles; 4 per wave
    #pragma unroll
    for (int i = 0; i < 4; ++i) {
        const int tid  = wave * 4 + i;
        const int mt   = tid >> 4;     // 0..1
        const int nt   = tid & 15;     // 0..15
        const int row  = mt * 16 + mlo;
        const int ncol = nt * 16 + mlo;
        v8f acc = {};
        for (int kk = 0; kk < 8; ++kk) {
            v16h a;
            const _Float16* ap = &A_s[row * LDA + kk * 32 + kbA];
            #pragma unroll
            for (int e = 0; e < 8; ++e) { a[e] = ap[e]; a[8 + e] = ap[8 + e]; }
            const v8h* bp = (const v8h*)&W1h[ncol * 256 + kk * 32 + kbB];
            const v8h b0 = bp[0], b1v = bp[1];
            v16h bf;
            #pragma unroll
            for (int e = 0; e < 8; ++e) { bf[e] = b0[e]; bf[8 + e] = b1v[e]; }
            acc = __builtin_amdgcn_wmma_f32_16x16x32_f16(
                false, a, false, bf, (short)0, acc, false, false);
        }
        const float bias = b1[ncol];
        #pragma unroll
        for (int v = 0; v < 8; ++v) {
            float hv = acc[v] + bias;
            hv = hv > 0.f ? hv : 0.f;
            const int m = mt * 16 + v + 8 * hi;
            H_s[m * LDA + ncol] = (_Float16)hv;
        }
    }
    __syncthreads();

    // ---- GEMM2: (MB x 256) x (256 x 128) + b2 -> Z_s (f32); B from LDS ----
    // 2 Mtiles x 8 Ntiles = 16 WMMA tiles; 2 per wave
    #pragma unroll
    for (int i = 0; i < 2; ++i) {
        const int tid  = wave * 2 + i;
        const int mt   = tid >> 3;     // 0..1
        const int nt   = tid & 7;      // 0..7
        const int row  = mt * 16 + mlo;
        const int ncol = nt * 16 + mlo;
        v8f acc = {};
        for (int kk = 0; kk < 8; ++kk) {
            v16h a;
            const _Float16* ap = &H_s[row * LDA + kk * 32 + kbA];
            #pragma unroll
            for (int e = 0; e < 8; ++e) { a[e] = ap[e]; a[8 + e] = ap[8 + e]; }
            const v8h* bp = (const v8h*)&W2_s[ncol * 256 + kk * 32 + kbB];
            const v8h b0 = bp[0], b1v = bp[1];
            v16h bf;
            #pragma unroll
            for (int e = 0; e < 8; ++e) { bf[e] = b0[e]; bf[8 + e] = b1v[e]; }
            acc = __builtin_amdgcn_wmma_f32_16x16x32_f16(
                false, a, false, bf, (short)0, acc, false, false);
        }
        const float bias = b2[ncol];
        #pragma unroll
        for (int v = 0; v < 8; ++v) {
            const int m = mt * 16 + v + 8 * hi;
            Z_s[m * 128 + ncol] = acc[v] + bias;
        }
    }
    __syncthreads();

    // ---- L2 norms over P=128 per row ----
    if (t < MB) {
        float s = 0.f;
        #pragma unroll 4
        for (int c = 0; c < 128; ++c) { float v = Z_s[t * 128 + c]; s += v * v; }
        rnorm_s[t] = 1.0f / fmaxf(sqrtf(s), 1e-12f);
    }
    __syncthreads();

    // ---- write normalized z, 16B per lane (b128 stores) ----
    for (int i = t; i < MB * 32; i += 256) {
        const int m  = i >> 5;
        const int c4 = i & 31;
        const float4 v = *(const float4*)&Z_s[m * 128 + c4 * 4];
        const float r = rnorm_s[m];
        float4 o; o.x = v.x * r; o.y = v.y * r; o.z = v.z * r; o.w = v.w * r;
        *(float4*)&z[((size_t)(base + m) << 7) + c4 * 4] = o;
    }
}

// ---------------------------------------------------------------------------
// Sampler: per batch, classify positions into hard/easy for class 1/0,
// deterministic first-N selection (stand-in for JAX random permutation),
// gather anchor(100)/pos(300)/neg(500) rows, zero-pad invalid rows.
// out = [anchor (800,128) | pos (2400,128) | neg (4000,128)]
// ---------------------------------------------------------------------------
__global__ __launch_bounds__(256) void sample_kernel(
    const float* __restrict__ z,
    const int* __restrict__ labels,
    const int* __restrict__ predict,
    float* __restrict__ out)
{
    __shared__ int h1[64];    // need <= 50
    __shared__ int e1[352];   // need <= nek1 + npk1 <= 350
    __shared__ int h0[112];   // need <= 100
    __shared__ int e0[512];   // need <= 500
    __shared__ int cnt[4];

    const int b = blockIdx.x;
    const int t = threadIdx.x;

    if (t == 0) {
        int nh1 = 0, ne1 = 0, nh0 = 0, ne0 = 0;
        const int* lb = labels + b * HWSZ;
        const int* pr = predict + b * HWSZ;
        for (int i = 0; i < HWSZ; ++i) {
            const int l = lb[i], p = pr[i];
            if (l == 1) {
                if (p != 1) { if (nh1 < 64)  h1[nh1] = i; nh1++; }
                else        { if (ne1 < 352) e1[ne1] = i; ne1++; }
            } else {
                if (p != 0) { if (nh0 < 112) h0[nh0] = i; nh0++; }
                else        { if (ne0 < 512) e0[ne0] = i; ne0++; }
            }
        }
        cnt[0] = nh1; cnt[1] = ne1; cnt[2] = nh0; cnt[3] = ne0;
    }
    __syncthreads();

    const int nh1 = cnt[0], ne1c = cnt[1], nh0 = cnt[2], ne0c = cnt[3];
    const int nhk1 = min(nh1, 50);
    const int nek1 = min(ne1c / 3, 50);
    const int npk1 = min(ne1c - nek1, 300);
    const int nhk0 = min(nh0, 100);
    const int nek0 = min(ne0c, 500 - nhk0);

    const float* zb = z + (size_t)b * HWSZ * 128;

    // anchors
    float* abase = out + (size_t)b * 100 * 128;
    for (int rc = t; rc < 100 * 128; rc += 256) {
        const int ja = rc >> 7, c = rc & 127;
        float v = 0.f;
        if (ja < nhk1 + nek1) {
            const int idx = (ja < nhk1) ? h1[ja] : e1[ja - nhk1];
            v = zb[(size_t)idx * 128 + c];
        }
        abase[rc] = v;
    }
    // positives
    float* pbase = out + (size_t)800 * 128 + (size_t)b * 300 * 128;
    for (int rc = t; rc < 300 * 128; rc += 256) {
        const int jp = rc >> 7, c = rc & 127;
        float v = 0.f;
        if (jp < npk1) {
            v = zb[(size_t)e1[nek1 + jp] * 128 + c];
        }
        pbase[rc] = v;
    }
    // negatives
    float* nbase = out + (size_t)(800 + 2400) * 128 + (size_t)b * 500 * 128;
    for (int rc = t; rc < 500 * 128; rc += 256) {
        const int jn = rc >> 7, c = rc & 127;
        float v = 0.f;
        if (jn < nhk0 + nek0) {
            const int idx = (jn < nhk0) ? h0[jn] : e0[jn - nhk0];
            v = zb[(size_t)idx * 128 + c];
        }
        nbase[rc] = v;
    }
}

extern "C" void kernel_launch(void* const* d_in, const int* in_sizes, int n_in,
                              void* d_out, int out_size, void* d_ws, size_t ws_size,
                              hipStream_t stream) {
    const float* x       = (const float*)d_in[0];
    const int*   labels  = (const int*)d_in[1];
    const int*   predict = (const int*)d_in[2];
    const float* W1      = (const float*)d_in[3];
    const float* b1      = (const float*)d_in[4];
    const float* W2      = (const float*)d_in[5];
    const float* b2      = (const float*)d_in[6];
    float* out = (float*)d_out;

    // ws layout: z (131072*128 f32 = 64MB) | W1h (128KB f16) | W2h (64KB f16)
    float*    zbuf = (float*)d_ws;
    _Float16* W1h  = (_Float16*)((char*)d_ws + (size_t)131072 * 128 * 4);
    _Float16* W2h  = W1h + 256 * 256;

    cvt_kernel<<<256, 256, 0, stream>>>(W1, W2, W1h, W2h);
    const int npos = 8 * HWSZ;   // 131072
    proj_kernel<<<npos / MB, 256, 0, stream>>>(x, W1h, b1, W2h, b2, zbuf);
    sample_kernel<<<8, 256, 0, stream>>>(zbuf, labels, predict, out);
}
